// SelfAttention_25185688223892
// MI455X (gfx1250) — matrix operand order repaired and checked by reading, not compiled, not run on hardware
//
#include <hip/hip_runtime.h>
#include <math.h>

#define HEADS    8
#define HEAD_DIM 32
#define N_TOK    4096
#define BATCH    2
#define BG       (BATCH * HEADS)
#define SCALE_F  0.17677669529663687f   // 32^-0.5

typedef __attribute__((ext_vector_type(16))) _Float16 v16h;
typedef __attribute__((ext_vector_type(8)))  _Float16 v8h;
typedef __attribute__((ext_vector_type(8)))  float    v8f;

// ---------------------------------------------------------------------------
// Kernel 1: grouped 1x1-conv projections Q,K,V (f32 math, f16 outputs).
// Q,K stored [bg][n][d] (row-major over d) for A/B fragment loads.
// V stored [bg][d][n] (row-major over n) for V A-fragment loads.
// Scale d^-1/2 folded into Q.
// ---------------------------------------------------------------------------
__global__ __launch_bounds__(256) void qkv_proj_kernel(
    const float* __restrict__ x,
    const float* __restrict__ Wq, const float* __restrict__ bq,
    const float* __restrict__ Wk, const float* __restrict__ bk,
    const float* __restrict__ Wv, const float* __restrict__ bv,
    _Float16* __restrict__ Qh, _Float16* __restrict__ Kh, _Float16* __restrict__ Vh)
{
    __shared__ float wq[32 * 32], wk[32 * 32], wv[32 * 32];
    __shared__ float biasq[32], biask[32], biasv[32];

    const int bg = blockIdx.y;
    const int b  = bg >> 3;
    const int g  = bg & 7;
    const int t  = threadIdx.x;

    #pragma unroll
    for (int r = 0; r < 4; ++r) {
        const int lin = r * 256 + t;
        wq[lin] = Wq[g * 1024 + lin];
        wk[lin] = Wk[g * 1024 + lin];
        wv[lin] = Wv[g * 1024 + lin];
    }
    if (t < 32) {
        biasq[t] = bq[g * 32 + t];
        biask[t] = bk[g * 32 + t];
        biasv[t] = bv[g * 32 + t];
    }
    __syncthreads();

    const int n = blockIdx.x * 256 + t;
    const float* xp = x + ((size_t)(b * 256 + g * 32)) * N_TOK + n;
    float xv[32];
    #pragma unroll
    for (int d = 0; d < 32; ++d) xv[d] = xp[(size_t)d * N_TOK];

    float acc[32];

    // ---- Q (scale folded in) ----
    #pragma unroll
    for (int o = 0; o < 32; ++o) {
        float s = biasq[o];
        #pragma unroll
        for (int d = 0; d < 32; ++d) s += wq[o * 32 + d] * xv[d];
        acc[o] = s * SCALE_F;
    }
    {
        _Float16* qrow = Qh + ((size_t)bg * N_TOK + n) * 32;
        #pragma unroll
        for (int c = 0; c < 4; ++c) {
            v8h pk;
            #pragma unroll
            for (int e = 0; e < 8; ++e) pk[e] = (_Float16)acc[c * 8 + e];
            *(v8h*)(qrow + c * 8) = pk;
        }
    }

    // ---- K ----
    #pragma unroll
    for (int o = 0; o < 32; ++o) {
        float s = biask[o];
        #pragma unroll
        for (int d = 0; d < 32; ++d) s += wk[o * 32 + d] * xv[d];
        acc[o] = s;
    }
    {
        _Float16* krow = Kh + ((size_t)bg * N_TOK + n) * 32;
        #pragma unroll
        for (int c = 0; c < 4; ++c) {
            v8h pk;
            #pragma unroll
            for (int e = 0; e < 8; ++e) pk[e] = (_Float16)acc[c * 8 + e];
            *(v8h*)(krow + c * 8) = pk;
        }
    }

    // ---- V (transposed layout: [d][n]) ----
    #pragma unroll
    for (int o = 0; o < 32; ++o) {
        float s = biasv[o];
        #pragma unroll
        for (int d = 0; d < 32; ++d) s += wv[o * 32 + d] * xv[d];
        Vh[((size_t)bg * 32 + o) * N_TOK + n] = (_Float16)s;
    }
}

// ---------------------------------------------------------------------------
// Kernel 2: flash attention + fused output projection.
// Grid: (32 i-tiles of 128 rows, 16 bg). 256 threads = 8 waves, 16 rows/wave.
// S^T = K*Q per 16x16 tile => softmax state is a per-lane scalar.
// O accumulated transposed (M=d, N=i) with A=V, B=P^T.
//
// Contraction-index relabeling for the PV matmul:
//   k=0..7 <-> j 0..7, k=8..15 <-> j 16..23, k=16..23 <-> j 8..15,
//   k=24..31 <-> j 24..31.
// With this labeling the P^T B-fragment is exactly the lanes' resident
// C-layout data (no cross-lane exchange), and each V A-fragment is two
// 16-byte v8h loads, 16 j apart, joined into a v16h.
// ---------------------------------------------------------------------------
__global__ __launch_bounds__(256) void flash_attn_kernel(
    const _Float16* __restrict__ Qh, const _Float16* __restrict__ Kh,
    const _Float16* __restrict__ Vh,
    const float* __restrict__ Wp, const float* __restrict__ bp,
    float* __restrict__ out)
{
    __shared__ float ldsO[128][32];
    __shared__ float ldsW[32 * 32];
    __shared__ float ldsB[32];

    const int bg   = blockIdx.y;
    const int b    = bg >> 3;
    const int g    = bg & 7;
    const int tile = blockIdx.x;
    const int t    = threadIdx.x;
    const int wave = t >> 5;
    const int lane = t & 31;
    const bool hiHalf = (lane >= 16);
    const int li   = lane & 15;

    #pragma unroll
    for (int r = 0; r < 4; ++r) ldsW[r * 256 + t] = Wp[g * 1024 + r * 256 + t];
    if (t < 32) ldsB[t] = bp[g * 32 + t];

    const int i0 = tile * 128 + wave * 16;

    // Q B-fragment: lane holds column i = i0+li, K(=d) = hiHalf ? 16..31 : 0..15
    const _Float16* qptr = Qh + ((size_t)bg * N_TOK + (i0 + li)) * 32 + (hiHalf ? 16 : 0);
    const v16h qB = *(const v16h*)qptr;

    v8f o0 = {};   // O^T tile, d = 0..15
    v8f o1 = {};   // O^T tile, d = 16..31
    float m_i = -INFINITY, l_i = 0.0f;

    const _Float16* Kbase = Kh + (size_t)bg * N_TOK * 32;
    const _Float16* Vbase = Vh + (size_t)bg * 32 * N_TOK;
    const int dOff = hiHalf ? 16 : 0;   // K A-fragment d-subset per lane half
    const int jOff = hiHalf ? 8 : 0;    // V A-fragment j-subset per lane half

    #pragma unroll 2
    for (int j0 = 0; j0 < N_TOK; j0 += 32) {
        // ---- K A-fragments (rows j, K=d): lane row = j0 (+16) + li ----
        const _Float16* kr0 = Kbase + (size_t)(j0 + li) * 32;
        const _Float16* kr1 = Kbase + (size_t)(j0 + 16 + li) * 32;
        const v8h k0lo = *(const v8h*)(kr0 + dOff);
        const v8h k0hi = *(const v8h*)(kr0 + dOff + 8);
        const v8h k1lo = *(const v8h*)(kr1 + dOff);
        const v8h k1hi = *(const v8h*)(kr1 + dOff + 8);
        // ---- V A-fragments (rows d, K=j): two v8h at j0+jOff and j0+jOff+16 ----
        const _Float16* vr0 = Vbase + (size_t)li * N_TOK + j0 + jOff;
        const _Float16* vr1 = Vbase + (size_t)(16 + li) * N_TOK + j0 + jOff;
        const v16h vA0 = __builtin_shufflevector(*(const v8h*)vr0, *(const v8h*)(vr0 + 16), 0, 1, 2, 3, 4, 5, 6, 7, 8, 9, 10, 11, 12, 13, 14, 15);
        const v16h vA1 = __builtin_shufflevector(*(const v8h*)vr1, *(const v8h*)(vr1 + 16), 0, 1, 2, 3, 4, 5, 6, 7, 8, 9, 10, 11, 12, 13, 14, 15);

        v16h kA0, kA1;
        #pragma unroll
        for (int e = 0; e < 8; ++e) {
            kA0[e] = k0lo[e]; kA0[8 + e] = k0hi[e];
            kA1[e] = k1lo[e]; kA1[8 + e] = k1hi[e];
        }

        // ---- S^T tiles: (j x i) = K(j,d) x Q(d,i), full d=32 in one WMMA ----
        const v8f zeroc = {};
        v8f st0 = __builtin_amdgcn_wmma_f32_16x16x32_f16(false, kA0, false, qB,
                                                         (short)0, zeroc, false, false);
        v8f st1 = __builtin_amdgcn_wmma_f32_16x16x32_f16(false, kA1, false, qB,
                                                         (short)0, zeroc, false, false);

        // ---- online softmax over this 32-key chunk (per-lane column i) ----
        float cmax = st0[0];
        #pragma unroll
        for (int v = 1; v < 8; ++v) cmax = fmaxf(cmax, st0[v]);
        #pragma unroll
        for (int v = 0; v < 8; ++v) cmax = fmaxf(cmax, st1[v]);
        cmax = fmaxf(cmax, __shfl_xor(cmax, 16, 32));

        const float mnew  = fmaxf(m_i, cmax);
        const float alpha = __expf(m_i - mnew);
        float p0[8], p1[8], psum = 0.0f;
        #pragma unroll
        for (int v = 0; v < 8; ++v) {
            p0[v] = __expf(st0[v] - mnew);
            p1[v] = __expf(st1[v] - mnew);
            psum += p0[v] + p1[v];
        }
        psum += __shfl_xor(psum, 16, 32);
        l_i = l_i * alpha + psum;
        m_i = mnew;
        #pragma unroll
        for (int v = 0; v < 8; ++v) { o0[v] *= alpha; o1[v] *= alpha; }

        // ---- P^T B-fragment: resident data, no cross-lane exchange ----
        v16h pB;
        #pragma unroll
        for (int e = 0; e < 8; ++e) {
            pB[e]     = (_Float16)p0[e];
            pB[8 + e] = (_Float16)p1[e];
        }

        o0 = __builtin_amdgcn_wmma_f32_16x16x32_f16(false, vA0, false, pB,
                                                    (short)0, o0, false, false);
        o1 = __builtin_amdgcn_wmma_f32_16x16x32_f16(false, vA1, false, pB,
                                                    (short)0, o1, false, false);

        if (j0 + 32 < N_TOK) {
            __builtin_prefetch((const void*)(kr0 + 32 * 32), 0, 1);  // next K chunk
            __builtin_prefetch((const void*)(Vbase + (size_t)li * N_TOK + j0 + 32),
                               0, 1);                                // next V chunk
        }
    }

    // ---- normalize, stage O^T -> LDS as [i][d] ----
    const float inv_l = 1.0f / l_i;
    const int iloc = wave * 16 + li;
    #pragma unroll
    for (int v = 0; v < 8; ++v) {
        ldsO[iloc][v + (hiHalf ? 8 : 0)]      = o0[v] * inv_l;
        ldsO[iloc][16 + v + (hiHalf ? 8 : 0)] = o1[v] * inv_l;
    }
    __syncthreads();

    // ---- fused output projection: out[o,i] = Wp[o,:] . O[:,i] + bp[o] ----
    const int i_l   = t >> 1;
    const int obase = (t & 1) * 16;
    float xr[32];
    #pragma unroll
    for (int d = 0; d < 32; ++d) xr[d] = ldsO[i_l][d];
    const size_t nglob = (size_t)tile * 128 + i_l;
    #pragma unroll
    for (int oo = 0; oo < 16; ++oo) {
        const int o = obase + oo;
        float s = ldsB[o];
        #pragma unroll
        for (int d = 0; d < 32; ++d) s += ldsW[o * 32 + d] * xr[d];
        out[((size_t)b * 256 + g * 32 + o) * N_TOK + nglob] = s;
    }
}

// ---------------------------------------------------------------------------
extern "C" void kernel_launch(void* const* d_in, const int* in_sizes, int n_in,
                              void* d_out, int out_size, void* d_ws, size_t ws_size,
                              hipStream_t stream)
{
    (void)in_sizes; (void)n_in; (void)out_size; (void)ws_size;

    const float* x  = (const float*)d_in[0];
    const float* Wq = (const float*)d_in[1];
    const float* bq = (const float*)d_in[2];
    const float* Wk = (const float*)d_in[3];
    const float* bk = (const float*)d_in[4];
    const float* Wv = (const float*)d_in[5];
    const float* bv = (const float*)d_in[6];
    const float* Wp = (const float*)d_in[7];
    const float* bp = (const float*)d_in[8];
    float* out = (float*)d_out;

    char* ws = (char*)d_ws;
    const size_t qkvBytes = (size_t)BG * N_TOK * 32 * sizeof(_Float16);  // 4 MB each
    _Float16* Qh = (_Float16*)(ws);
    _Float16* Kh = (_Float16*)(ws + qkvBytes);
    _Float16* Vh = (_Float16*)(ws + 2 * qkvBytes);

    qkv_proj_kernel<<<dim3(16, 16), 256, 0, stream>>>(
        x, Wq, bq, Wk, bk, Wv, bv, Qh, Kh, Vh);
    flash_attn_kernel<<<dim3(32, 16), 256, 0, stream>>>(
        Qh, Kh, Vh, Wp, bp, out);
}
